// BidirectionalCrossAttentionBlock_16372415332872
// MI455X (gfx1250) — matrix operand order repaired and checked by reading, not compiled, not run on hardware
//
#include <hip/hip_runtime.h>
#include <hip/hip_bf16.h>

typedef __attribute__((ext_vector_type(16))) __bf16 v16bf;
typedef __attribute__((ext_vector_type(8)))  float  v8f;

#define FEAT 256
#define GUIDE 512
#define HW 4096
#define NB 16
#define ATTN_SCALE 0.125f   // (256/4)^-0.5

// ---------------------------------------------------------------------------
// K0: fp32 -> bf16 weight conversion for the two FFN GEMMs
// ---------------------------------------------------------------------------
__global__ void k_cvt_w(const float* __restrict__ w1, const float* __restrict__ w2,
                        __bf16* __restrict__ w1b, __bf16* __restrict__ w2b) {
    int i = blockIdx.x * 256 + threadIdx.x;
    w1b[i] = (__bf16)w1[i];
    w2b[i] = (__bf16)w2[i];
}

// ---------------------------------------------------------------------------
// K1: text path (tiny). Per batch: LN over D=512, q_txt/v_txt projections,
// tgate sigmoid, and qk[b,:] = q_txt^T * W_img_k (folds the k_img GEMM away).
// One block per batch, 256 threads.
// ---------------------------------------------------------------------------
__global__ void k_txt(const float* __restrict__ txt,
                      const float* __restrict__ g, const float* __restrict__ be,
                      const float* __restrict__ wq, const float* __restrict__ wv,
                      const float* __restrict__ wtg, const float* __restrict__ btg,
                      const float* __restrict__ wik,
                      float* __restrict__ vtxt, float* __restrict__ qk,
                      float* __restrict__ timp) {
    __shared__ float tn[GUIDE];
    __shared__ float red[256];
    __shared__ float qs[FEAT];
    int b = blockIdx.x, t = threadIdx.x;
    float x0 = txt[b * GUIDE + t];
    float x1 = txt[b * GUIDE + 256 + t];

    red[t] = x0 + x1;
    __syncthreads();
    for (int s = 128; s > 0; s >>= 1) { if (t < s) red[t] += red[t + s]; __syncthreads(); }
    float m = red[0] * (1.0f / GUIDE);
    __syncthreads();
    red[t] = (x0 - m) * (x0 - m) + (x1 - m) * (x1 - m);
    __syncthreads();
    for (int s = 128; s > 0; s >>= 1) { if (t < s) red[t] += red[t + s]; __syncthreads(); }
    float var = red[0] * (1.0f / GUIDE);
    float rs = rsqrtf(var + 1e-5f);

    float t0 = (x0 - m) * rs * g[t]       + be[t];
    float t1 = (x1 - m) * rs * g[256 + t] + be[256 + t];
    tn[t] = t0; tn[256 + t] = t1;
    __syncthreads();

    // tgate reduction
    red[t] = t0 * wtg[t] + t1 * wtg[256 + t];
    __syncthreads();
    for (int s = 128; s > 0; s >>= 1) { if (t < s) red[t] += red[t + s]; __syncthreads(); }
    if (t == 0) timp[b] = 1.0f / (1.0f + expf(-(red[0] + btg[0])));

    // q_txt / v_txt projections (each thread one output channel)
    float q = 0.0f, vv = 0.0f;
    for (int d = 0; d < GUIDE; ++d) {
        float tv = tn[d];
        q  += tv * wq[t * GUIDE + d];
        vv += tv * wv[t * GUIDE + d];
    }
    qs[t] = q;
    vtxt[b * FEAT + t] = vv;
    __syncthreads();

    // qk[c'] = sum_o q[o] * W_img_k[o, c']  (column access, coalesced over t)
    float s = 0.0f;
    for (int o = 0; o < FEAT; ++o) s += qs[o] * wik[o * FEAT + t];
    qk[b * FEAT + t] = s;
}

// ---------------------------------------------------------------------------
// K2: per-pixel channel LayerNorm fused with igate dot and attention-logit
// dot. Writes img_n (bf16), img_imp, raw scaled logits. One thread per pixel.
// ---------------------------------------------------------------------------
__global__ void k_lnimg(const float* __restrict__ img,
                        const float* __restrict__ g, const float* __restrict__ be,
                        const float* __restrict__ qk,
                        const float* __restrict__ wig, const float* __restrict__ big,
                        __bf16* __restrict__ imgn, float* __restrict__ imp,
                        float* __restrict__ logits) {
    int pix = blockIdx.x * 256 + threadIdx.x;      // 0 .. NB*HW-1
    int b = pix >> 12, n = pix & (HW - 1);
    size_t base = (size_t)b * FEAT * HW + n;

    float s = 0.0f, ss = 0.0f;
    for (int c = 0; c < FEAT; ++c) {
        float x = img[base + (size_t)c * HW];
        s += x; ss += x * x;
    }
    float m  = s * (1.0f / FEAT);
    float var = ss * (1.0f / FEAT) - m * m;
    float rs = rsqrtf(var + 1e-5f);

    const float* qkb = qk + b * FEAT;
    float gd = 0.0f, ld = 0.0f;
    for (int c = 0; c < FEAT; ++c) {
        float x  = img[base + (size_t)c * HW];
        float xn = (x - m) * rs * g[c] + be[c];
        imgn[base + (size_t)c * HW] = (__bf16)xn;
        gd += wig[c] * xn;
        ld += qkb[c] * xn;
    }
    imp[pix]    = 1.0f / (1.0f + expf(-(gd + big[0])));
    logits[pix] = ld * ATTN_SCALE;
}

// ---------------------------------------------------------------------------
// K3: softmax over N=4096 per batch (in-place logits -> attn). One block / b.
// ---------------------------------------------------------------------------
__global__ void k_softmax(float* __restrict__ logits) {
    __shared__ float red[256];
    int b = blockIdx.x, t = threadIdx.x;
    float l[16];
    float mx = -3.0e38f;
#pragma unroll
    for (int i = 0; i < 16; ++i) {
        l[i] = logits[b * HW + t + i * 256];
        mx = fmaxf(mx, l[i]);
    }
    red[t] = mx; __syncthreads();
    for (int s = 128; s > 0; s >>= 1) { if (t < s) red[t] = fmaxf(red[t], red[t + s]); __syncthreads(); }
    mx = red[0]; __syncthreads();
    float sum = 0.0f;
#pragma unroll
    for (int i = 0; i < 16; ++i) { l[i] = expf(l[i] - mx); sum += l[i]; }
    red[t] = sum; __syncthreads();
    for (int s = 128; s > 0; s >>= 1) { if (t < s) red[t] += red[t + s]; __syncthreads(); }
    float inv = 1.0f / red[0];
#pragma unroll
    for (int i = 0; i < 16; ++i) logits[b * HW + t + i * 256] = l[i] * inv;
}

// ---------------------------------------------------------------------------
// K4: s[b,c] = sum_n attn[b,n] * img_n[b,c,n]. One wave32 per (b,c) row.
// ---------------------------------------------------------------------------
__global__ void k_sreduce(const __bf16* __restrict__ imgn, const float* __restrict__ attn,
                          float* __restrict__ sout) {
    int row  = blockIdx.x * 8 + (threadIdx.x >> 5);   // b*FEAT + c
    int lane = threadIdx.x & 31;
    int b = row >> 8;
    size_t base = (size_t)row * HW;
    float acc = 0.0f;
    for (int n = lane; n < HW; n += 32)
        acc += attn[b * HW + n] * (float)imgn[base + n];
#pragma unroll
    for (int off = 16; off > 0; off >>= 1) acc += __shfl_xor(acc, off, 32);
    if (lane == 0) sout[row] = acc;
}

// ---------------------------------------------------------------------------
// K5: tiny per-batch matvecs: tg = W_img_v * s ; wt = W_out * tg ; wv = W_out * v_txt
// ---------------------------------------------------------------------------
__global__ void k_proj(const float* __restrict__ s, const float* __restrict__ vtxt,
                       const float* __restrict__ wiv, const float* __restrict__ wout,
                       float* __restrict__ wt, float* __restrict__ wv) {
    __shared__ float sv[FEAT], vvs[FEAT], tg[FEAT];
    int b = blockIdx.x, t = threadIdx.x;
    sv[t]  = s[b * FEAT + t];
    vvs[t] = vtxt[b * FEAT + t];
    __syncthreads();
    float a = 0.0f;
    for (int c = 0; c < FEAT; ++c) a += wiv[t * FEAT + c] * sv[c];
    tg[t] = a;
    __syncthreads();
    float aw = 0.0f, av = 0.0f;
    for (int o = 0; o < FEAT; ++o) {
        float w = wout[t * FEAT + o];
        aw += w * tg[o];
        av += w * vvs[o];
    }
    wt[b * FEAT + t] = aw;
    wv[b * FEAT + t] = av;
}

// ---------------------------------------------------------------------------
// K6: out1 = img + b_out + (ii*wv + ti*wt)/(ii + ti + eps)   (folded W_out conv)
// ---------------------------------------------------------------------------
__global__ void k_fuse(const float* __restrict__ img, const float* __restrict__ imp,
                       const float* __restrict__ wv, const float* __restrict__ wt,
                       const float* __restrict__ timp, const float* __restrict__ bout,
                       float* __restrict__ out1) {
    size_t i = (size_t)blockIdx.x * 256 + threadIdx.x;
    int n = (int)(i & (HW - 1));
    int c = (int)((i >> 12) & 255);
    int b = (int)(i >> 20);
    float ii = imp[(b << 12) | n];
    float ti = timp[b];
    float num = ii * wv[(b << 8) | c] + ti * wt[(b << 8) | c];
    out1[i] = img[i] + bout[c] + num / (ii + ti + 1e-8f);
}

// ---------------------------------------------------------------------------
// WMMA fragment loaders (per CDNA5 ISA 16-bit A 16x32 / B 32x16 VGPR layouts)
// ---------------------------------------------------------------------------
#define LDA 40
#define LDB 40

__device__ inline v16bf load_a_frag(const __bf16* sA, int mBase, int lane) {
    union { unsigned int u[8]; v16bf v; } r;
    int row  = mBase + (lane & 15);
    int half = lane >> 4;
    const unsigned int* p = (const unsigned int*)(sA + row * LDA);
#pragma unroll
    for (int i = 0; i < 4; ++i) r.u[i]     = p[half * 4 + i];       // K = half*8 + 2i
#pragma unroll
    for (int i = 0; i < 4; ++i) r.u[4 + i] = p[8 + half * 4 + i];   // K = 16 + half*8 + 2i
    return r.v;
}

__device__ inline v16bf load_b_frag(const __bf16* sB, int nBase, int lane) {
    union { unsigned int u[8]; v16bf v; } r;
    int col  = nBase + (lane & 15);
    int half = lane >> 4;
    const unsigned int* p = (const unsigned int*)(sB + col * LDB);  // [n][k] layout
#pragma unroll
    for (int j = 0; j < 8; ++j) r.u[j] = p[(j >> 2) * 8 + half * 4 + (j & 3)];  // K = (j/4)*16 + half*8 + 2(j%4)
    return r.v;
}

#define WMMA_BF16(A, B, C) \
    __builtin_amdgcn_wmma_f32_16x16x32_bf16(false, (A), false, (B), (short)0, (C), false, false)

// ---------------------------------------------------------------------------
// K7/K8: FFN GEMM, y = W(256x256) @ x(256 x 4096) per batch, bf16 WMMA f32 acc.
// Block tile 128(M) x 64(N), 8 waves each computing a 32x32 tile (2x2 WMMA).
// MODE 0: x = out1 (f32, cvt->bf16), epilogue bias + exact GELU -> bf16 h
// MODE 1: x = h (bf16),              epilogue bias + residual   -> f32 out
// ---------------------------------------------------------------------------
template <int MODE>
__global__ void k_ffn(const __bf16* __restrict__ Wb, const float* __restrict__ bias,
                      const float* __restrict__ inF, const __bf16* __restrict__ inH,
                      const float* __restrict__ res, float* __restrict__ outF,
                      __bf16* __restrict__ outH) {
    __shared__ __bf16 sA[128 * LDA];
    __shared__ __bf16 sB[64 * LDB];

    int ct = blockIdx.x;
    int b  = ct >> 6;
    int n0 = (ct & 63) << 6;
    int m0 = blockIdx.y << 7;
    int t = threadIdx.x, lane = t & 31, wave = t >> 5;
    int wm = wave >> 1, wn = wave & 1;

    v8f c00 = {}, c01 = {}, c10 = {}, c11 = {};

    int arow = t >> 1, ahalf = t & 1;   // A staging: 128 rows x 32 cols, 16 bf16/thread
    int brow = t >> 3, bseg = t & 7;    // B staging: 32 k-rows x 64 cols, 8 elems/thread
    size_t batchBase = (size_t)b * FEAT * HW;

    for (int k0 = 0; k0 < FEAT; k0 += 32) {
        // ---- stage A: weight tile (bf16, 16B-aligned vector loads) ----
        const int4* ga = (const int4*)(Wb + (size_t)(m0 + arow) * FEAT + k0 + ahalf * 16);
        int4 av0 = ga[0];
        int4 av1 = ga[1];
        *(int4*)&sA[arow * LDA + ahalf * 16]     = av0;
        *(int4*)&sA[arow * LDA + ahalf * 16 + 8] = av1;

        // ---- stage B: activation tile into [n][k] layout ----
        if (MODE == 0) {
            const float* gb = inF + batchBase + (size_t)(k0 + brow) * HW + n0 + bseg * 8;
            float4 f0 = *(const float4*)gb;
            float4 f1 = *(const float4*)(gb + 4);
            __bf16 tmp[8] = {(__bf16)f0.x, (__bf16)f0.y, (__bf16)f0.z, (__bf16)f0.w,
                             (__bf16)f1.x, (__bf16)f1.y, (__bf16)f1.z, (__bf16)f1.w};
#pragma unroll
            for (int j = 0; j < 8; ++j) sB[(bseg * 8 + j) * LDB + brow] = tmp[j];
        } else {
            const __bf16* gb = inH + batchBase + (size_t)(k0 + brow) * HW + n0 + bseg * 8;
            int4 hv = *(const int4*)gb;
            __bf16 tmp[8];
            *(int4*)tmp = hv;
#pragma unroll
            for (int j = 0; j < 8; ++j) sB[(bseg * 8 + j) * LDB + brow] = tmp[j];
        }
        __syncthreads();

        v16bf a0 = load_a_frag(sA, wm * 32,      lane);
        v16bf a1 = load_a_frag(sA, wm * 32 + 16, lane);
        v16bf b0 = load_b_frag(sB, wn * 32,      lane);
        v16bf b1 = load_b_frag(sB, wn * 32 + 16, lane);

        c00 = WMMA_BF16(a0, b0, c00);
        c01 = WMMA_BF16(a0, b1, c01);
        c10 = WMMA_BF16(a1, b0, c10);
        c11 = WMMA_BF16(a1, b1, c11);
        __syncthreads();
    }

    // ---- epilogue: C/D layout: VGPR r, lanes0-15 -> M=r, lanes16-31 -> M=r+8 ----
    int colL = lane & 15, rhalf = lane >> 4;
    int waveM = m0 + wm * 32, waveN = n0 + wn * 32;
    v8f accs[4] = {c00, c01, c10, c11};
#pragma unroll
    for (int q = 0; q < 4; ++q) {
        int msub = (q >> 1) * 16, nsub = (q & 1) * 16;
        int nglob = waveN + nsub + colL;
        v8f a = accs[q];
#pragma unroll
        for (int r = 0; r < 8; ++r) {
            int mglob = waveM + msub + r + rhalf * 8;
            size_t idx = batchBase + (size_t)mglob * HW + nglob;
            float val = a[r] + bias[mglob];
            if (MODE == 0) {
                float gl = 0.5f * val * (1.0f + erff(val * 0.70710678118f));
                outH[idx] = (__bf16)gl;
            } else {
                outF[idx] = val + res[idx];
            }
        }
    }
}

// ---------------------------------------------------------------------------
extern "C" void kernel_launch(void* const* d_in, const int* in_sizes, int n_in,
                              void* d_out, int out_size, void* d_ws, size_t ws_size,
                              hipStream_t stream) {
    (void)in_sizes; (void)n_in; (void)out_size; (void)ws_size;
    const float* img  = (const float*)d_in[0];
    const float* txt  = (const float*)d_in[1];
    const float* lig  = (const float*)d_in[2];
    const float* lib  = (const float*)d_in[3];
    const float* ltg  = (const float*)d_in[4];
    const float* ltb  = (const float*)d_in[5];
    // d_in[6] w_img_q: dead (softmax over singleton axis == 1)
    const float* wik  = (const float*)d_in[7];
    const float* wiv  = (const float*)d_in[8];
    const float* wtq  = (const float*)d_in[9];
    // d_in[10] w_txt_k: dead
    const float* wtv  = (const float*)d_in[11];
    const float* wtg  = (const float*)d_in[12];
    const float* btg  = (const float*)d_in[13];
    const float* wig  = (const float*)d_in[14];
    const float* big  = (const float*)d_in[15];
    const float* wout = (const float*)d_in[16];
    const float* bout = (const float*)d_in[17];
    const float* w1   = (const float*)d_in[18];
    const float* b1   = (const float*)d_in[19];
    const float* w2   = (const float*)d_in[20];
    const float* b2   = (const float*)d_in[21];
    float* outp = (float*)d_out;

    char* ws = (char*)d_ws;
    size_t off = 0;
    auto alloc = [&](size_t bytes) -> char* {
        char* p = ws + off;
        off = (off + bytes + 255) & ~(size_t)255;
        return p;
    };
    constexpr size_t NELEM = (size_t)NB * FEAT * HW;     // 16.7M
    __bf16* imgn  = (__bf16*)alloc(NELEM * 2);           // 32 MB; reused as h after k_sreduce
    float* out1   = (float*)alloc(NELEM * 4);            // 64 MB
    float* logits = (float*)alloc((size_t)NB * HW * 4);  // becomes attn in-place
    float* imp    = (float*)alloc((size_t)NB * HW * 4);
    float* qk     = (float*)alloc((size_t)NB * FEAT * 4);
    float* vtxt   = (float*)alloc((size_t)NB * FEAT * 4);
    float* sred   = (float*)alloc((size_t)NB * FEAT * 4);
    float* wtb    = (float*)alloc((size_t)NB * FEAT * 4);
    float* wvb    = (float*)alloc((size_t)NB * FEAT * 4);
    float* timp   = (float*)alloc((size_t)NB * 4);
    __bf16* w1b   = (__bf16*)alloc((size_t)FEAT * FEAT * 2);
    __bf16* w2b   = (__bf16*)alloc((size_t)FEAT * FEAT * 2);
    __bf16* hbuf  = imgn;  // alias: imgn is dead after k_sreduce

    k_cvt_w  <<<dim3(FEAT * FEAT / 256), dim3(256), 0, stream>>>(w1, w2, w1b, w2b);
    k_txt    <<<dim3(NB),                dim3(256), 0, stream>>>(txt, ltg, ltb, wtq, wtv, wtg, btg, wik, vtxt, qk, timp);
    k_lnimg  <<<dim3(NB * HW / 256),     dim3(256), 0, stream>>>(img, lig, lib, qk, wig, big, imgn, imp, logits);
    k_softmax<<<dim3(NB),                dim3(256), 0, stream>>>(logits);
    k_sreduce<<<dim3(NB * FEAT / 8),     dim3(256), 0, stream>>>(imgn, logits, sred);
    k_proj   <<<dim3(NB),                dim3(256), 0, stream>>>(sred, vtxt, wiv, wout, wtb, wvb);
    k_fuse   <<<dim3((unsigned)(NELEM / 256)), dim3(256), 0, stream>>>(img, imp, wvb, wtb, timp, bout, out1);
    k_ffn<0> <<<dim3(NB * HW / 64, FEAT / 128), dim3(256), 0, stream>>>(w1b, b1, out1, nullptr, nullptr, nullptr, hbuf);
    k_ffn<1> <<<dim3(NB * HW / 64, FEAT / 128), dim3(256), 0, stream>>>(w2b, b2, nullptr, hbuf, out1, outp, nullptr);
}
